// TranslatorModel_8117488190123
// MI455X (gfx1250) — matrix operand order repaired and checked by reading, not compiled, not run on hardware
//
#include <hip/hip_runtime.h>
#include <hip/hip_bf16.h>
#include <math.h>

// ---------------------------------------------------------------------------
// Seq2seq LSTM translator for MI455X (gfx1250, wave32).
//
//  1. bf16 weight copies (W_hr_y bf16 = 51MB -> L2-resident on 192MB L2).
//  2. Teacher forcing => precompute Gx = X @ W_ih^T + b_ih + b_hh with a
//     parallel WMMA GEMM (decoder activations padded to 144 rows so the
//     GEMM is completely guard-free).
//  3. Sequential LSTM step = ONE fused kernel: 32 waves x 4 gate tiles of
//     h @ W_hh^T via WMMA (h in row 0 of A, select-zeroed, no EXEC branches),
//     gates staged in LDS, pointwise i/f/g/o update after s_barrier.
//  4. Decoder projection: 3125 WMMA 16-row tiles over W_hr_y with
//     global_prefetch_b8 on the next K-slice; single-workgroup LDS
//     log-softmax / CE reduction per step.
// ---------------------------------------------------------------------------

#define EDIM     512
#define G4E      2048        // 4*EDIM
#define S_LEN    128
#define T_LEN    128
#define DEC_ROWS 144         // T_LEN+1 padded to a multiple of 16
#define VT       50000
#define EOS_ID   1

typedef __bf16 bf16_t;
typedef __attribute__((ext_vector_type(16))) __bf16 v16bf;
typedef __attribute__((ext_vector_type(8)))  __bf16 v8bf;
typedef __attribute__((ext_vector_type(8)))  float  v8f;

#define WMMA_BF16(a, b, c) \
  __builtin_amdgcn_wmma_f32_16x16x32_bf16(false, (a), false, (b), (short)0, (c), false, false)

// ------------------------- elementwise helpers ------------------------------

__global__ void k_f32_to_bf16(const float* __restrict__ src,
                              bf16_t* __restrict__ dst, int n) {
  int i = blockIdx.x * blockDim.x + threadIdx.x;
  if (i < n) dst[i] = (bf16_t)src[i];
}

// Xenc[t] = W_x_hi[source_words[t]]                         (S_LEN rows)
// Xdec[0] = W_x_hi[EOS]; Xdec[r] = W_y_hi[target_words[r-1]] (r=1..T_LEN)
// Xdec rows T_LEN+1 .. DEC_ROWS-1 are zero padding (guard-free GEMM).
__global__ void k_gather(const int* __restrict__ sw, const int* __restrict__ tw,
                         const float* __restrict__ Wx, const float* __restrict__ Wy,
                         bf16_t* __restrict__ Xenc, bf16_t* __restrict__ Xdec) {
  int idx = blockIdx.x * blockDim.x + threadIdx.x;
  int total = (S_LEN + DEC_ROWS) * EDIM;
  if (idx >= total) return;
  int row = idx / EDIM, col = idx % EDIM;
  if (row < S_LEN) {
    Xenc[(size_t)row * EDIM + col] = (bf16_t)Wx[(size_t)sw[row] * EDIM + col];
  } else {
    int r = row - S_LEN;
    float v = 0.0f;
    if (r == 0)           v = Wx[(size_t)EOS_ID * EDIM + col];
    else if (r <= T_LEN)  v = Wy[(size_t)tw[r - 1] * EDIM + col];
    Xdec[(size_t)r * EDIM + col] = (bf16_t)v;
  }
}

__global__ void k_init_state(bf16_t* hbf, float* c, float* acc) {
  int i = threadIdx.x;
  if (i < EDIM) { hbf[i] = (bf16_t)0.0f; c[i] = 0.0f; }
  if (i == 0) acc[0] = 0.0f;
}

__global__ void k_reset_c(float* c) {
  int i = threadIdx.x;
  if (i < EDIM) c[i] = 0.0f;
}

// --------------------------- WMMA tile loads --------------------------------
// A (16x32 bf16, MxK): lane = m + 16*h holds K in {h*16..h*16+15}.
// Guard-free version (caller guarantees all 16 rows are readable).
__device__ __forceinline__ v16bf load_a_tile(const bf16_t* __restrict__ base,
                                             int ld, int m0, int k0, int lane) {
  int m = m0 + (lane & 15);
  int hh = lane >> 4;
  const bf16_t* row = base + (size_t)m * ld + k0 + hh * 16;
  v8bf lo = *(const v8bf*)(row);
  v8bf hi = *(const v8bf*)(row + 8);
  v16bf a;
#pragma unroll
  for (int i = 0; i < 8; ++i) { a[i] = lo[i]; a[8 + i] = hi[i]; }
  return a;
}

// Matvec A tile: h in row 0, rows 1..15 zero. All lanes load (broadcast 32B
// from L0), non-row-0 lanes zeroed via selects -> no EXEC-mask branches.
__device__ __forceinline__ v16bf load_h_tile(const bf16_t* __restrict__ h,
                                             int k0, int lane) {
  int hh = lane >> 4;
  const bf16_t* p = h + k0 + hh * 16;
  v8bf lo = *(const v8bf*)(p);
  v8bf hi = *(const v8bf*)(p + 8);
  bool r0 = (lane & 15) == 0;
  bf16_t z = (bf16_t)0.0f;
  v16bf a;
#pragma unroll
  for (int i = 0; i < 8; ++i) {
    a[i]     = r0 ? lo[i] : z;
    a[8 + i] = r0 ? hi[i] : z;
  }
  return a;
}

// B (32x16 bf16, KxN) from row-major W[j,k]: lane = n + 16*h holds column n,
// K in {h*16 .. h*16+15} -> one contiguous 32B load per lane.
__device__ __forceinline__ v16bf load_b_tile(const bf16_t* __restrict__ W,
                                             int ld, int j0, int k0, int lane) {
  int n = lane & 15;
  int hh = lane >> 4;
  return *(const v16bf*)(W + (size_t)(j0 + n) * ld + k0 + hh * 16);
}

// ------------------------------ GEMM: X @ W_ih^T ----------------------------
// Gx[m, j] = sum_k X[m,k]*Wih[j,k] + b_ih[j] + b_hh[j]; one 16x16 tile/wave.
// M is always a multiple of 16 (decoder padded) -> no guards anywhere.
__global__ void k_gemm_ih(const bf16_t* __restrict__ X,
                          const bf16_t* __restrict__ Wih,
                          const float* __restrict__ b_ih,
                          const float* __restrict__ b_hh,
                          float* __restrict__ Gx, int Mtiles) {
  int lane = threadIdx.x & 31;
  int wave = blockIdx.x * (blockDim.x >> 5) + (threadIdx.x >> 5);
  int ntiles = G4E / 16;
  if (wave >= Mtiles * ntiles) return;
  int mt = wave / ntiles;
  int nt = wave % ntiles;
  int j0 = nt * 16;
  v8f c = {};
#pragma unroll
  for (int kk = 0; kk < EDIM / 32; ++kk) {
    v16bf a = load_a_tile(X, EDIM, mt * 16, kk * 32, lane);
    v16bf b = load_b_tile(Wih, EDIM, j0, kk * 32, lane);
    c = WMMA_BF16(a, b, c);
  }
  int n = lane & 15, hh = lane >> 4;
  float bias = b_ih[j0 + n] + b_hh[j0 + n];
#pragma unroll
  for (int r = 0; r < 8; ++r) {
    int m = mt * 16 + hh * 8 + r;   // D layout: VGPR r = row r (+8 hi half)
    Gx[(size_t)m * G4E + j0 + n] = c[r] + bias;
  }
}

// ------------------- fused sequential LSTM step (1 WG) ----------------------
// 32 waves; wave w computes gates j in [w*64, w*64+64) = 4 WMMA tiles of
// h @ W_hh^T, adds Gx row, stages gates in LDS, then the pointwise i/f/g/o
// update runs after the workgroup barrier. hbf is read pre-barrier and
// written post-barrier, so in-place update is safe.
__global__ __launch_bounds__(1024) void k_lstm_step(
    bf16_t* __restrict__ hbf, const bf16_t* __restrict__ Whh,
    const float* __restrict__ GxRow, float* __restrict__ cvec) {
  __shared__ float sg[G4E];
  int lane = threadIdx.x & 31;
  int wave = threadIdx.x >> 5;       // 0..31
  int jbase = wave * 64;

  v8f c0 = {}, c1 = {}, c2 = {}, c3 = {};
#pragma unroll
  for (int kk = 0; kk < EDIM / 32; ++kk) {
    v16bf a  = load_h_tile(hbf, kk * 32, lane);
    v16bf b0 = load_b_tile(Whh, EDIM, jbase,      kk * 32, lane);
    c0 = WMMA_BF16(a, b0, c0);
    v16bf b1 = load_b_tile(Whh, EDIM, jbase + 16, kk * 32, lane);
    c1 = WMMA_BF16(a, b1, c1);
    v16bf b2 = load_b_tile(Whh, EDIM, jbase + 32, kk * 32, lane);
    c2 = WMMA_BF16(a, b2, c2);
    v16bf b3 = load_b_tile(Whh, EDIM, jbase + 48, kk * 32, lane);
    c3 = WMMA_BF16(a, b3, c3);
  }
  if (lane < 16) {                   // row 0 of D = VGPR 0, lanes 0..15
    sg[jbase + lane]      = GxRow[jbase + lane]      + c0[0];
    sg[jbase + 16 + lane] = GxRow[jbase + 16 + lane] + c1[0];
    sg[jbase + 32 + lane] = GxRow[jbase + 32 + lane] + c2[0];
    sg[jbase + 48 + lane] = GxRow[jbase + 48 + lane] + c3[0];
  }
  __syncthreads();

  int i = threadIdx.x;               // torch gate order i, f, g, o
  if (i < EDIM) {
    float gi = sg[i],            gf = sg[EDIM + i];
    float gg = sg[2 * EDIM + i], go = sg[3 * EDIM + i];
    float si = 1.0f / (1.0f + __expf(-gi));
    float sf = 1.0f / (1.0f + __expf(-gf));
    float so = 1.0f / (1.0f + __expf(-go));
    float cn = sf * cvec[i] + si * tanhf(gg);
    cvec[i] = cn;
    hbf[i] = (bf16_t)(so * tanhf(cn));
  }
}

// ----------------------- decoder output projection --------------------------
// logits[v] = b_hr[v] + sum_k Whr[v,k]*h[k]; 3125 tiles of 16 rows, with
// global_prefetch of each lane's next K-slice of the L2-resident weights.
__global__ void k_proj(const bf16_t* __restrict__ hbf,
                       const bf16_t* __restrict__ Whr,
                       const float* __restrict__ b_hr,
                       float* __restrict__ logits) {
  int lane = threadIdx.x & 31;
  int wave = blockIdx.x * (blockDim.x >> 5) + (threadIdx.x >> 5);
  if (wave >= VT / 16) return;
  int j0 = wave * 16;
  int n = lane & 15, hh = lane >> 4;
  const bf16_t* wrow = Whr + (size_t)(j0 + n) * EDIM + hh * 16;
  v8f c = {};
#pragma unroll
  for (int kk = 0; kk < EDIM / 32; ++kk) {
    if (kk + 1 < EDIM / 32)
      __builtin_prefetch(wrow + (kk + 1) * 32, 0, 0);  // global_prefetch_b8
    v16bf a = load_h_tile(hbf, kk * 32, lane);
    v16bf b = *(const v16bf*)(wrow + kk * 32);
    c = WMMA_BF16(a, b, c);
  }
  if (hh == 0) logits[j0 + n] = c[0] + b_hr[j0 + n];
}

// --------------------- log-softmax cross-entropy loss -----------------------
__global__ void k_loss(const float* __restrict__ logits,
                       const int* __restrict__ tw, int step,
                       float* __restrict__ acc, float* __restrict__ out) {
  __shared__ float sred[1024];
  int tid = threadIdx.x;
  float m = -INFINITY;
  for (int v = tid; v < VT; v += 1024) m = fmaxf(m, logits[v]);
  sred[tid] = m;
  __syncthreads();
  for (int s = 512; s > 0; s >>= 1) {
    if (tid < s) sred[tid] = fmaxf(sred[tid], sred[tid + s]);
    __syncthreads();
  }
  float mv = sred[0];
  __syncthreads();
  float se = 0.0f;
  for (int v = tid; v < VT; v += 1024) se += __expf(logits[v] - mv);
  sred[tid] = se;
  __syncthreads();
  for (int s = 512; s > 0; s >>= 1) {
    if (tid < s) sred[tid] += sred[tid + s];
    __syncthreads();
  }
  if (tid == 0) {
    int tok = (step < T_LEN) ? tw[step] : EOS_ID;
    float loss = (mv + __logf(sred[0])) - logits[tok];
    float a = acc[0] + loss;
    acc[0] = a;
    out[0] = a;   // last decoder step leaves the final summed loss
  }
}

// ------------------------------ host driver ---------------------------------

extern "C" void kernel_launch(void* const* d_in, const int* in_sizes, int n_in,
                              void* d_out, int out_size, void* d_ws, size_t ws_size,
                              hipStream_t stream) {
  const int*   sw      = (const int*)d_in[0];
  const int*   tw      = (const int*)d_in[1];
  const float* Wx      = (const float*)d_in[2];
  const float* Wy      = (const float*)d_in[3];
  const float* encWih  = (const float*)d_in[4];
  const float* encWhh  = (const float*)d_in[5];
  const float* enc_bih = (const float*)d_in[6];
  const float* enc_bhh = (const float*)d_in[7];
  const float* decWih  = (const float*)d_in[8];
  const float* decWhh  = (const float*)d_in[9];
  const float* dec_bih = (const float*)d_in[10];
  const float* dec_bhh = (const float*)d_in[11];
  const float* Whr     = (const float*)d_in[12];
  const float* bhr     = (const float*)d_in[13];
  float* out = (float*)d_out;

  char* p = (char*)d_ws;
  auto alloc = [&](size_t bytes) -> void* {
    void* r = (void*)p;
    p += (bytes + 255) & ~(size_t)255;
    return r;
  };

  bf16_t* encWih_b = (bf16_t*)alloc((size_t)G4E * EDIM * sizeof(bf16_t));
  bf16_t* encWhh_b = (bf16_t*)alloc((size_t)G4E * EDIM * sizeof(bf16_t));
  bf16_t* decWih_b = (bf16_t*)alloc((size_t)G4E * EDIM * sizeof(bf16_t));
  bf16_t* decWhh_b = (bf16_t*)alloc((size_t)G4E * EDIM * sizeof(bf16_t));
  bf16_t* Whr_b    = (bf16_t*)alloc((size_t)VT * EDIM * sizeof(bf16_t));
  bf16_t* Xenc     = (bf16_t*)alloc((size_t)S_LEN * EDIM * sizeof(bf16_t));
  bf16_t* Xdec     = (bf16_t*)alloc((size_t)DEC_ROWS * EDIM * sizeof(bf16_t));
  float*  GxEnc    = (float*)alloc((size_t)S_LEN * G4E * sizeof(float));
  float*  GxDec    = (float*)alloc((size_t)DEC_ROWS * G4E * sizeof(float));
  bf16_t* hbf      = (bf16_t*)alloc((size_t)EDIM * sizeof(bf16_t));
  float*  cvec     = (float*)alloc((size_t)EDIM * sizeof(float));
  float*  logits   = (float*)alloc((size_t)VT * sizeof(float));
  float*  acc      = (float*)alloc(sizeof(float));

  // 1. bf16 weight conversion (one pass; stays hot in L2 thereafter)
  {
    int n = G4E * EDIM;
    int blk = (n + 255) / 256;
    k_f32_to_bf16<<<blk, 256, 0, stream>>>(encWih, encWih_b, n);
    k_f32_to_bf16<<<blk, 256, 0, stream>>>(encWhh, encWhh_b, n);
    k_f32_to_bf16<<<blk, 256, 0, stream>>>(decWih, decWih_b, n);
    k_f32_to_bf16<<<blk, 256, 0, stream>>>(decWhh, decWhh_b, n);
    int nh = VT * EDIM;
    k_f32_to_bf16<<<(nh + 255) / 256, 256, 0, stream>>>(Whr, Whr_b, nh);
  }

  // 2. gather embeddings (zero-padded decoder rows), init h/c/accumulator
  {
    int total = (S_LEN + DEC_ROWS) * EDIM;
    k_gather<<<(total + 255) / 256, 256, 0, stream>>>(sw, tw, Wx, Wy, Xenc, Xdec);
    k_init_state<<<1, 512, 0, stream>>>(hbf, cvec, acc);
  }

  // 3. precompute input-gate GEMMs (parallel WMMA; biases folded in)
  {
    int MtE = S_LEN / 16;                               // 8
    int wavesE = MtE * (G4E / 16);                      // 1024
    k_gemm_ih<<<(wavesE + 7) / 8, 256, 0, stream>>>(Xenc, encWih_b, enc_bih,
                                                    enc_bhh, GxEnc, MtE);
    int MtD = DEC_ROWS / 16;                            // 9
    int wavesD = MtD * (G4E / 16);                      // 1152
    k_gemm_ih<<<(wavesD + 7) / 8, 256, 0, stream>>>(Xdec, decWih_b, dec_bih,
                                                    dec_bhh, GxDec, MtD);
  }

  const int projBlocks = (VT / 16 + 7) / 8;             // 391 blocks x 8 waves

  // 4. encoder recurrence (one fused kernel per step)
  for (int t = 0; t < S_LEN; ++t) {
    k_lstm_step<<<1, 1024, 0, stream>>>(hbf, encWhh_b,
                                        GxEnc + (size_t)t * G4E, cvec);
  }

  // cell state reset before decoding (as in the reference)
  k_reset_c<<<1, 512, 0, stream>>>(cvec);

  // 5. decoder recurrence + projection + CE loss
  for (int t = 0; t < T_LEN + 1; ++t) {
    k_lstm_step<<<1, 1024, 0, stream>>>(hbf, decWhh_b,
                                        GxDec + (size_t)t * G4E, cvec);
    k_proj<<<projBlocks, 256, 0, stream>>>(hbf, Whr_b, bhr, logits);
    k_loss<<<1, 1024, 0, stream>>>(logits, tw, t, acc, out);
  }
}
